// ProbabilisticFractionalLayer_20925080666467
// MI455X (gfx1250) — matrix operand order repaired and checked by reading, not compiled, not run on hardware
//
#include <hip/hip_runtime.h>
#include <hip/hip_bf16.h>
#include <cstdint>
#include <cstddef>

#define NN  65536
#define BBA 16
#define KK  64
#define TPB 256

typedef _Float16     v16h __attribute__((ext_vector_type(16)));
typedef float        v8f  __attribute__((ext_vector_type(8)));
typedef unsigned int v4u  __attribute__((ext_vector_type(4)));
typedef int          v8i  __attribute__((ext_vector_type(8)));
typedef int          v4i  __attribute__((ext_vector_type(4)));

__device__ __forceinline__ _Float16 bits_to_h(unsigned short u) {
  union { unsigned short u; _Float16 h; } cvt; cvt.u = u; return cvt.h;
}
__device__ __forceinline__ unsigned short h_to_bits(_Float16 h) {
  union { unsigned short u; _Float16 h; } cvt; cvt.h = h; return cvt.u;
}

// ---------------------------------------------------------------------------
// Prep: alpha, c, lag table j[64], weights w[64] (f32) and packed f16 hi/lo
// split of w for compensated WMMA accumulation. Layout in d_ws (dwords):
//   [0..63]    j (int32)
//   [64..127]  w (f32)
//   [128..191] packed {lo16 = f16(w), hi16 = f16(w - f16(w))}
//   [192]      c (f32)
// ---------------------------------------------------------------------------
__global__ void frac_prep(const float* __restrict__ loc,
                          const float* __restrict__ scale,
                          const float* __restrict__ eps,
                          const int* __restrict__ lags,
                          uint32_t* __restrict__ ws) {
  const int m = threadIdx.x;
  const float sp = log1pf(expf(scale[0]));          // softplus
  float alpha = loc[0] + sp * eps[0];
  alpha = fminf(fmaxf(alpha, 0.01f), 0.99f);
  if (m < KK) {
    int j = lags[m];
    j = min(max(j, 1), NN - 1);
    const float w = powf((float)j, -(alpha + 1.0f));
    const _Float16 wh = (_Float16)w;
    const _Float16 wl = (_Float16)(w - (float)wh);
    ((int*)ws)[m]        = j;
    ((float*)ws)[64 + m] = w;
    ws[128 + m] = (uint32_t)h_to_bits(wh) | ((uint32_t)h_to_bits(wl) << 16);
  }
  if (m == 0) {
    const float c = alpha * expf(-lgammaf(1.0f - alpha)) * (float)(NN - 1) / (float)KK;
    ((float*)ws)[192] = c;
  }
}

// ---------------------------------------------------------------------------
// Main: block = 256 threads = 256 consecutive t for one batch row b.
//  1) TDM tensor_load_to_lds stages the 192-dword table into LDS.
//  2) Each wave computes S[t] = sum_m mask(t,m)*w_m for its 32 t's via
//     v_wmma_f32_16x16x32_f16 (mask is exact 0/1 in f16; w split hi+lo).
//  3) Per-thread predicated-FMA gather loop for sum_m mask*w_m*x[b,t-j_m].
// ---------------------------------------------------------------------------
__global__ __launch_bounds__(TPB) void frac_main(const float* __restrict__ x,
                                                 float* __restrict__ out,
                                                 const uint32_t* __restrict__ ws) {
  __shared__ uint32_t sTable[192];
  __shared__ float    sS[TPB];

  const int tid  = threadIdx.x;
  const int b    = blockIdx.y;
  const int t0   = blockIdx.x * TPB;
  const int wave = tid >> 5;
  const int lane = tid & 31;
  const int g    = (lane >> 4) & 1;   // 16-lane group

#if __has_builtin(__builtin_amdgcn_tensor_load_to_lds) && __has_builtin(__builtin_amdgcn_s_wait_tensorcnt)
  if (wave == 0) {
    const unsigned long long ga = (unsigned long long)(uintptr_t)ws;
    const unsigned int ldsoff   = (unsigned int)(uintptr_t)(&sTable[0]);
    // D# group0: count=1, lds_addr, global_addr[56:0], type=2 ("image")
    v4u g0 = { 1u, ldsoff, (unsigned int)(ga & 0xFFFFFFFFull),
               ((unsigned int)((ga >> 32) & 0x1FFFFFFull)) | 0x80000000u };
    // D# group1: data_size=4B; tensor_dim0=192, tensor_dim1=1; tile_dim0=192,
    // tile_dim1=1; tensor_dim0_stride=192, tensor_dim1_stride=192.
    v8i g1 = { 0x00020000, (int)(192u << 16), (int)(1u << 16), (int)(192u << 16),
               1, 192, (int)(192u << 16), 0 };
    v4i gz = { 0, 0, 0, 0 };
#if __clang_major__ >= 23
    v8i gz8 = { 0, 0, 0, 0, 0, 0, 0, 0 };
    __builtin_amdgcn_tensor_load_to_lds(g0, g1, gz, gz, gz8, 0);
#else
    __builtin_amdgcn_tensor_load_to_lds(g0, g1, gz, gz, 0);
#endif
    __builtin_amdgcn_s_wait_tensorcnt(0);
  }
#else
  for (int i = tid; i < 192; i += TPB) sTable[i] = ws[i];
#endif
  __syncthreads();

  const int*      sJ = (const int*)sTable;
  const float*    sW = (const float*)(sTable + 64);
  const uint32_t* sP = sTable + 128;

  // --- S[t] via WMMA: D[m,n] = sum_k A[m,k]*B[k,n], A = mask tile (16 t-rows
  // x 32 lags), B[k,n] = w_k broadcast over n; every D column equals S.
  for (int htile = 0; htile < 2; ++htile) {
    const int tbase = t0 + wave * 32 + htile * 16;
    const int trow  = tbase + (lane & 15);          // A-matrix row M = lane%16
    v8f acc = { 0.f, 0.f, 0.f, 0.f, 0.f, 0.f, 0.f, 0.f };
    for (int kc = 0; kc < KK; kc += 32) {
      v16h A, Bh, Bl;
#pragma unroll
      for (int h = 0; h < 16; ++h) {
        const int v = h >> 1, o = h & 1;
        // A 16x32 f16 layout (matches B): lanes0-15 k=0..15, lanes16-31 k=16..31
        const int ka = kc + 2 * v + o + 16 * g;
        A[h] = (trow >= sJ[ka]) ? (_Float16)1.0f : (_Float16)0.0f;
        // B 32x16 f16 layout: lanes0-15 k=0..15, lanes16-31 k=16..31
        const int kb = kc + 2 * v + o + 16 * g;
        const uint32_t p = sP[kb];
        Bh[h] = bits_to_h((unsigned short)(p & 0xFFFFu));
        Bl[h] = bits_to_h((unsigned short)(p >> 16));
      }
      acc = __builtin_amdgcn_wmma_f32_16x16x32_f16(false, A, false, Bh,
                                                   (short)0, acc, false, false);
      acc = __builtin_amdgcn_wmma_f32_16x16x32_f16(false, A, false, Bl,
                                                   (short)0, acc, false, false);
    }
    // Column n=0 lives in lanes 0 (rows 0..7) and 16 (rows 8..15).
    if ((lane & 15) == 0) {
      const int mbase = g * 8;
#pragma unroll
      for (int r = 0; r < 8; ++r) sS[tbase - t0 + mbase + r] = acc[r];
    }
  }
  __syncthreads();

  // --- gather reduction: coalesced per-lag streams, all hits in L2 (x = 4MB)
  const float c   = ((const float*)ws)[192];
  const int   t   = t0 + tid;
  const float* xr = x + (size_t)b * NN;
  const float xt  = xr[t];
  float accv = 0.0f;
#pragma unroll 4
  for (int m = 0; m < KK; ++m) {
    const int j = sJ[m];
    if (t >= j) accv = fmaf(xr[t - j], sW[m], accv);
  }
  out[(size_t)b * NN + t] = c * (xt * sS[tid] - accv);
}

extern "C" void kernel_launch(void* const* d_in, const int* in_sizes, int n_in,
                              void* d_out, int out_size, void* d_ws, size_t ws_size,
                              hipStream_t stream) {
  const float* x     = (const float*)d_in[0];
  const float* loc   = (const float*)d_in[1];
  const float* scale = (const float*)d_in[2];
  const float* eps   = (const float*)d_in[3];
  const int*   lags  = (const int*)d_in[4];
  float*       out   = (float*)d_out;
  uint32_t*    ws    = (uint32_t*)d_ws;

  frac_prep<<<dim3(1), dim3(64), 0, stream>>>(loc, scale, eps, lags, ws);
  frac_main<<<dim3(NN / TPB, BBA), dim3(TPB), 0, stream>>>(x, out, ws);
}
